// VRNNDecoder_18073222381758
// MI455X (gfx1250) — matrix operand order repaired and checked by reading, compiled, not run on hardware
//
#include <hip/hip_runtime.h>
#include <hip/hip_bf16.h>
#include <math.h>

// ---- problem constants (match reference) ----
#define INPUT   512
#define HIDDEN  1024
#define LATENT  512
#define GIN     1024          // INPUT + LATENT
#define BB      64
#define TT      512
#define MROWS   (BB * TT)     // 32768
#define CATD    1536          // HIDDEN + INPUT
#define G3H     (3 * HIDDEN)  // 3072

#define APAD    40            // LDS A-tile row stride in halves (32 + 8 pad)
#define WPAD    1032          // LDS W_hh row stride in halves (1024 + 8 pad)

typedef __bf16 v16bf __attribute__((ext_vector_type(16)));
typedef __bf16 v8bf  __attribute__((ext_vector_type(8)));
typedef float  v8f   __attribute__((ext_vector_type(8)));

static __device__ __forceinline__ unsigned short f32_to_bf16_rne(float f) {
  union { float f; unsigned u; } x; x.f = f;
  unsigned r = x.u + 0x7FFFu + ((x.u >> 16) & 1u);
  return (unsigned short)(r >> 16);
}

union FragBF { v16bf v; v8bf h[2]; };

// A fragment: 16x32 bf16 from row-major [*, lda] global memory.
// ISA: lanes 0-15 hold M=lane, halves {K0..7, K16..23}; lanes 16-31 hold {K8..15, K24..31}.
static __device__ __forceinline__ v16bf load_a_frag(const unsigned short* A, long lda,
                                                    int m0, int k0, int lane) {
  int m  = m0 + (lane & 15);
  int kb = k0 + ((lane & 16) ? 8 : 0);
  const unsigned short* p = A + (long)m * lda + kb;
  FragBF f;
  f.h[0] = *reinterpret_cast<const v8bf*>(p);
  f.h[1] = *reinterpret_cast<const v8bf*>(p + 16);
  return f.v;
}

// A fragment from an LDS-staged 64x32 tile (row stride APAD halves).
static __device__ __forceinline__ v16bf load_a_frag_lds(const unsigned short* sA,
                                                        int m0, int lane) {
  int m  = m0 + (lane & 15);
  int kb = (lane & 16) ? 8 : 0;
  const unsigned short* p = sA + m * APAD + kb;
  FragBF f;
  f.h[0] = *reinterpret_cast<const v8bf*>(p);
  f.h[1] = *reinterpret_cast<const v8bf*>(p + 16);
  return f.v;
}

// B fragment: 32x16 (KxN) sourced from W row-major [N, K]: B[k][n] = W[n0+n][k0+k].
// ISA: lanes 0-15 hold column n = lane with K={0..7,16..23}; lanes 16-31 hold K={8..15,24..31}.
static __device__ __forceinline__ v16bf load_b_frag(const unsigned short* W, long ldw,
                                                    int n0, int k0, int lane) {
  int n  = n0 + (lane & 15);
  int kb = k0 + ((lane & 16) ? 8 : 0);
  const unsigned short* p = W + (long)n * ldw + kb;
  FragBF f;
  f.h[0] = *reinterpret_cast<const v8bf*>(p);
  f.h[1] = *reinterpret_cast<const v8bf*>(p + 16);
  return f.v;
}

// B fragment from LDS-resident W_hh slice: row r = g*16 + n, row stride WPAD halves.
static __device__ __forceinline__ v16bf load_b_frag_lds(const unsigned short* sW,
                                                        int g, int k0, int lane) {
  int r  = g * 16 + (lane & 15);
  int kb = k0 + ((lane & 16) ? 8 : 0);
  const unsigned short* p = sW + r * WPAD + kb;
  FragBF f;
  f.h[0] = *reinterpret_cast<const v8bf*>(p);
  f.h[1] = *reinterpret_cast<const v8bf*>(p + 16);
  return f.v;
}

static __device__ __forceinline__ float sigm(float x) {
  return 1.0f / (1.0f + __expf(-x));
}

static __device__ __forceinline__ v8f wmma_bf16(v16bf a, v16bf b, v8f c) {
  return __builtin_amdgcn_wmma_f32_16x16x32_bf16(false, a, false, b, (short)0, c, false, false);
}

// ---------------- prep kernels ----------------
__global__ void cvt_f32_bf16(const float* __restrict__ src, unsigned short* __restrict__ dst, int n) {
  int i = blockIdx.x * blockDim.x + threadIdx.x;
  if (i < n) dst[i] = f32_to_bf16_rne(src[i]);
}

__global__ void prep_inp(const int* __restrict__ x, const float* __restrict__ z,
                         const float* __restrict__ emb,
                         unsigned short* __restrict__ inp,    // [MROWS, GIN]
                         unsigned short* __restrict__ cat) {  // [MROWS, CATD], emb part
  int r = blockIdx.x;              // r = b*T + t
  int b = r / TT;
  int tok = x[r];
  for (int i = threadIdx.x; i < INPUT; i += blockDim.x) {
    unsigned short e = f32_to_bf16_rne(emb[(long)tok * INPUT + i]);
    inp[(long)r * GIN + i] = e;
    cat[(long)r * CATD + HIDDEN + i] = e;
    inp[(long)r * GIN + INPUT + i] = f32_to_bf16_rne(z[(long)b * LATENT + i]);
  }
}

__global__ void zero_u32(unsigned* __restrict__ p, int n) {
  for (int i = blockIdx.x * blockDim.x + threadIdx.x; i < n; i += gridDim.x * blockDim.x)
    p[i] = 0u;
}

// ---------------- generic bf16 GEMM with LDS-staged A tile ----------------
// Block: 256 threads (8 waves). Block tile: 64 (M) x 256 (N). Wave tile: 64 x 32.
// A [MROWS, KD] staged per 32-K slab into LDS (double-buffered through registers);
// B fragments register-pipelined one K-iteration ahead.
template <int KD, bool DO_TANH>
__global__ void __launch_bounds__(256) gemm_bf16(const unsigned short* __restrict__ A,
                                                 const unsigned short* __restrict__ W,
                                                 const float* __restrict__ bias,
                                                 float* __restrict__ C, int ldc) {
  __shared__ unsigned short sA[64 * APAD];
  const int tid  = threadIdx.x;
  const int lane = tid & 31;
  const int wave = tid >> 5;
  const int m0 = blockIdx.x * 64;
  const int n0 = blockIdx.y * 256 + wave * 32;

  // cooperative A staging: thread -> (row = tid/4, 8-half chunk = (tid%4)*8)
  const int srow = tid >> 2;
  const int skp  = (tid & 3) * 8;
  const unsigned short* grow = A + (long)(m0 + srow) * KD + skp;
  unsigned short* sdst = sA + srow * APAD + skp;

  const v8f zero = {0,0,0,0,0,0,0,0};
  v8f acc[4][2] = {{zero,zero},{zero,zero},{zero,zero},{zero,zero}};

  // prologue: stage k0 = 0 into LDS, prefetch first B fragments
  *reinterpret_cast<v8bf*>(sdst) = *reinterpret_cast<const v8bf*>(grow);
  v16bf bfr0 = load_b_frag(W, KD, n0,      0, lane);
  v16bf bfr1 = load_b_frag(W, KD, n0 + 16, 0, lane);
  __syncthreads();

  for (int k0 = 0; k0 < KD; k0 += 32) {
    const bool more = (k0 + 32) < KD;
    v8bf nxt;
    v16bf b0n = bfr0, b1n = bfr1;
    if (more) {
      nxt = *reinterpret_cast<const v8bf*>(grow + k0 + 32);  // next A slab -> regs
      b0n = load_b_frag(W, KD, n0,      k0 + 32, lane);      // next B frags -> regs
      b1n = load_b_frag(W, KD, n0 + 16, k0 + 32, lane);
    }
#pragma unroll
    for (int mt = 0; mt < 4; ++mt) {
      v16bf af = load_a_frag_lds(sA, mt * 16, lane);
      acc[mt][0] = wmma_bf16(af, bfr0, acc[mt][0]);
      acc[mt][1] = wmma_bf16(af, bfr1, acc[mt][1]);
    }
    bfr0 = b0n; bfr1 = b1n;
    __syncthreads();
    if (more) *reinterpret_cast<v8bf*>(sdst) = nxt;
    __syncthreads();
  }

  const int nc0 = n0 + (lane & 15);
  const float bv0 = bias[nc0];
  const float bv1 = bias[nc0 + 16];
#pragma unroll
  for (int mt = 0; mt < 4; ++mt)
#pragma unroll
    for (int i = 0; i < 8; ++i) {
      int row = m0 + mt * 16 + i + ((lane & 16) ? 8 : 0);
      float v0 = acc[mt][0][i] + bv0;
      float v1 = acc[mt][1][i] + bv1;
      if (DO_TANH) { v0 = tanhf(v0); v1 = tanhf(v1); }
      C[(long)row * ldc + nc0]      = v0;
      C[(long)row * ldc + nc0 + 16] = v1;
    }
}

// ---------------- persistent GRU scan ----------------
// 64 WGs x 128 threads; each WG owns 16 hidden columns across all 3 gates and
// keeps its 48x1024 bf16 W_hh slice resident in LDS (~97 KB of the WGP's 320 KB)
// for all 512 steps. Inner K loop is software-pipelined one stage deep so the
// ds_load_b128 latency overlaps the XDL WMMAs. h is exchanged through L2 with
// agent-scope fences + a device-wide sense-reversing barrier.
__global__ void __launch_bounds__(128) gru_scan(const float* __restrict__ xproj,  // [MROWS, 3072]
                                                const float* __restrict__ bhh,    // [3072]
                                                const int* __restrict__ xtok,     // [B*T]
                                                const unsigned short* __restrict__ Whh, // [3072,1024] bf16
                                                float* __restrict__ h32,          // [2][B*H]
                                                unsigned short* __restrict__ hbf, // [2][B*H]
                                                unsigned short* __restrict__ cat, // [MROWS, CATD]
                                                unsigned* __restrict__ bar) {     // {counter, gen}
  extern __shared__ unsigned short sW[];    // [48][WPAD]
  const int lane = threadIdx.x & 31;
  const int wave = threadIdx.x >> 5;
  const int m0 = wave * 16;                 // batch rows handled by this wave
  const int j0 = blockIdx.x * 16;           // hidden columns handled by this WG
  const int j = j0 + (lane & 15);

  // one-time stage of the W_hh slice: rows r = g*16+n -> W_hh[g*H + j0 + n][:]
  for (int idx = threadIdx.x; idx < 48 * 64; idx += blockDim.x) {
    int r = idx >> 6;
    int c = (idx & 63) * 16;
    int g = r >> 4, n = r & 15;
    const unsigned short* src = Whh + (long)(g * HIDDEN + j0 + n) * HIDDEN + c;
    *reinterpret_cast<v8bf*>(sW + r * WPAD + c) = *reinterpret_cast<const v8bf*>(src);
  }
  __syncthreads();

  const float bh0 = bhh[j];
  const float bh1 = bhh[HIDDEN + j];
  const float bh2 = bhh[2 * HIDDEN + j];
  const v8f zero = {0,0,0,0,0,0,0,0};

  for (int t = 0; t < TT; ++t) {
    const int cur = t & 1;
    const unsigned short* hb = hbf + (long)cur * (BB * HIDDEN);
    const float* hc = h32 + (long)cur * (BB * HIDDEN);
    float* hn32 = h32 + (long)(cur ^ 1) * (BB * HIDDEN);
    unsigned short* hnbf = hbf + (long)(cur ^ 1) * (BB * HIDDEN);

    v8f acc[3] = {zero, zero, zero};

    // software-pipelined K loop: stage s holds fragments for iteration k0,
    // while loads for k0+32 are in flight during the three WMMAs.
    v16bf af  = load_a_frag(hb, HIDDEN, m0, 0, lane);
    v16bf bf0 = load_b_frag_lds(sW, 0, 0, lane);
    v16bf bf1 = load_b_frag_lds(sW, 1, 0, lane);
    v16bf bf2 = load_b_frag_lds(sW, 2, 0, lane);
#pragma unroll 4
    for (int k0 = 0; k0 < HIDDEN; k0 += 32) {
      v16bf afn = af, b0n = bf0, b1n = bf1, b2n = bf2;
      if (k0 + 32 < HIDDEN) {
        afn = load_a_frag(hb, HIDDEN, m0, k0 + 32, lane);
        b0n = load_b_frag_lds(sW, 0, k0 + 32, lane);
        b1n = load_b_frag_lds(sW, 1, k0 + 32, lane);
        b2n = load_b_frag_lds(sW, 2, k0 + 32, lane);
      }
      acc[0] = wmma_bf16(af, bf0, acc[0]);
      acc[1] = wmma_bf16(af, bf1, acc[1]);
      acc[2] = wmma_bf16(af, bf2, acc[2]);
      af = afn; bf0 = b0n; bf1 = b1n; bf2 = b2n;
    }

#pragma unroll
    for (int i = 0; i < 8; ++i) {
      const int b = m0 + i + ((lane & 16) ? 8 : 0);    // batch index
      const long row = (long)b * TT + t;
      const float xr = xproj[row * G3H + j];
      const float xu = xproj[row * G3H + HIDDEN + j];
      const float xn = xproj[row * G3H + 2 * HIDDEN + j];
      const float hp = hc[(long)b * HIDDEN + j];
      const float r = sigm(xr + acc[0][i] + bh0);
      const float u = sigm(xu + acc[1][i] + bh1);
      const float nv = tanhf(xn + r * (acc[2][i] + bh2));
      float hnew = (1.0f - u) * nv + u * hp;
      if (xtok[row] == 0) hnew = hp;                   // PAD mask freezes hidden
      hn32[(long)b * HIDDEN + j] = hnew;
      unsigned short hb16 = f32_to_bf16_rne(hnew);
      hnbf[(long)b * HIDDEN + j] = hb16;
      cat[row * CATD + j] = hb16;                      // record hidden for output GEMM
    }

    // device-wide sense-reversing barrier; agent-scope release/acquire fences
    // (global_wb / global_inv) make the new h visible across WGPs.
    __builtin_amdgcn_fence(__ATOMIC_RELEASE, "agent");
    __syncthreads();
    if (threadIdx.x == 0) {
      volatile unsigned* vgen = bar + 1;
      unsigned g = *vgen;
      unsigned arrived = atomicAdd(bar, 1u) + 1u;
      if (arrived == gridDim.x) {
        *((volatile unsigned*)bar) = 0u;
        __threadfence();
        *vgen = g + 1u;
      } else {
        while (*vgen == g) { __builtin_amdgcn_s_sleep(2); }
      }
    }
    __syncthreads();
    __builtin_amdgcn_fence(__ATOMIC_ACQUIRE, "agent");
  }
}

// ---------------- launch ----------------
extern "C" void kernel_launch(void* const* d_in, const int* in_sizes, int n_in,
                              void* d_out, int out_size, void* d_ws, size_t ws_size,
                              hipStream_t stream) {
  (void)in_sizes; (void)n_in; (void)out_size; (void)ws_size;
  const int*   x    = (const int*)  d_in[0];
  const float* z    = (const float*)d_in[1];
  const float* emb  = (const float*)d_in[2];
  const float* Wih  = (const float*)d_in[3];
  const float* bih  = (const float*)d_in[4];
  const float* Whh  = (const float*)d_in[5];
  const float* bhh  = (const float*)d_in[6];
  const float* Wout = (const float*)d_in[7];
  const float* bout = (const float*)d_in[8];

  char* ws = (char*)d_ws;
  size_t off = 0;
  auto alloc = [&](size_t bytes) -> void* {
    void* p = ws + off;
    off = (off + bytes + 255) & ~(size_t)255;
    return p;
  };
  unsigned short* Wih_b  = (unsigned short*)alloc((size_t)G3H  * GIN    * 2);
  unsigned short* Whh_b  = (unsigned short*)alloc((size_t)G3H  * HIDDEN * 2);
  unsigned short* Wout_b = (unsigned short*)alloc((size_t)INPUT* CATD   * 2);
  unsigned short* inp_b  = (unsigned short*)alloc((size_t)MROWS* GIN    * 2);
  unsigned short* cat_b  = (unsigned short*)alloc((size_t)MROWS* CATD   * 2);
  float*          xproj  = (float*)         alloc((size_t)MROWS* G3H    * 4);
  float*          h32    = (float*)         alloc((size_t)2 * BB * HIDDEN * 4);
  unsigned short* hbf    = (unsigned short*)alloc((size_t)2 * BB * HIDDEN * 2);
  unsigned*       bar    = (unsigned*)      alloc(256);

  // 1) weight conversion to bf16
  int nih = G3H * GIN, nhh = G3H * HIDDEN, nwo = INPUT * CATD;
  cvt_f32_bf16<<<(nih + 255) / 256, 256, 0, stream>>>(Wih,  Wih_b,  nih);
  cvt_f32_bf16<<<(nhh + 255) / 256, 256, 0, stream>>>(Whh,  Whh_b,  nhh);
  cvt_f32_bf16<<<(nwo + 255) / 256, 256, 0, stream>>>(Wout, Wout_b, nwo);

  // 2) embedding gather + [emb|z] pack (bf16), emb half of cat
  prep_inp<<<MROWS, 256, 0, stream>>>(x, z, emb, inp_b, cat_b);

  // 3) zero h (f32 + bf16, both phases) and barrier state (deterministic per call)
  zero_u32<<<(2 * BB * HIDDEN + 255) / 256, 256, 0, stream>>>((unsigned*)h32, 2 * BB * HIDDEN);
  zero_u32<<<(BB * HIDDEN + 255) / 256, 256, 0, stream>>>((unsigned*)hbf, BB * HIDDEN);
  zero_u32<<<1, 64, 0, stream>>>(bar, 64);

  // 4) x_proj GEMM: M=32768, N=3072, K=1024 (bf16 WMMA, f32 acc, +b_ih)
  gemm_bf16<GIN, false><<<dim3(MROWS / 64, G3H / 256), 256, 0, stream>>>(
      inp_b, Wih_b, bih, xproj, G3H);

  // 5) persistent GRU scan: 64 co-resident WGs, W_hh slice LDS-resident (~97 KB dyn-LDS)
  gru_scan<<<HIDDEN / 16, 128, (size_t)(48 * WPAD * 2), stream>>>(
      xproj, bhh, x, Whh_b, h32, hbf, cat_b, bar);

  // 6) output GEMM + tanh: M=32768, N=512, K=1536
  gemm_bf16<CATD, true><<<dim3(MROWS / 64, INPUT / 256), 256, 0, stream>>>(
      cat_b, Wout_b, bout, (float*)d_out, INPUT);
}
